// Cross_Attention_17669495456353
// MI455X (gfx1250) — matrix operand order repaired and checked by reading, not compiled, not run on hardware
//
#include <hip/hip_runtime.h>
#include <hip/hip_bf16.h>

// ---------------------------------------------------------------------------
// Cross-attention (3-branch AdaLN) for MI455X (gfx1250, wave32, WMMA).
// All heavy math on v_wmma_f32_16x16x32_f16 with f32 accumulation.
// Attention computed transposed (S^T = K*Q^T, O^T = V^T*P^T) so softmax
// statistics reduce in-register (1 shuffle per stat). Key-padding mask is
// precomputed as an additive f32 mask and fused into the logit FMA; softmax
// runs in the exp2 domain so each probability is a single v_exp_f32.
// ---------------------------------------------------------------------------

typedef __attribute__((ext_vector_type(16))) _Float16 v16h;
typedef __attribute__((ext_vector_type(8)))  _Float16 v8h;
typedef __attribute__((ext_vector_type(8)))  float    v8f;

#define BB  4
#define TT  512
#define NN  512
#define DD  1024
#define EE  1024
#define HH  16
#define HD  64
#define T3  1536   // 3*T

// 1/sqrt(HD) * log2(e): logits scaled straight into the exp2 domain
#define SCL2 0.18033688011112042f

__device__ __forceinline__ v16h cat8(v8h lo, v8h hi) {
    return __builtin_shufflevector(lo, hi, 0,1,2,3,4,5,6,7,8,9,10,11,12,13,14,15);
}
__device__ __forceinline__ v8f wmma16(v16h a, v16h b, v8f c) {
    // v_wmma_f32_16x16x32_f16: D = A(16x32) * B(32x16) + C
    return __builtin_amdgcn_wmma_f32_16x16x32_f16(false, a, false, b, (short)0, c, false, false);
}

// ---------------------------------------------------------------- SiLU(emb)
__global__ void silu_kernel(const float* __restrict__ emb, float* __restrict__ se) {
    int i = blockIdx.x * 256 + threadIdx.x;           // 4096 total
    float v = emb[i];
    se[i] = v / (1.f + __expf(-v));
}

// ---------------------- key_padding_mask (bool) -> additive f32 mask
__global__ void maskf_kernel(const unsigned char* __restrict__ mask,
                             float* __restrict__ mf) {
    int i = blockIdx.x * 256 + threadIdx.x;           // 2048 total
    mf[i] = mask[i] ? -1e30f : 0.f;
}

// --------------------------------------------- AdaLN GEMV: se @ W + b (tiny)
// scsh layout: [mat(4)][b(4)][2048] ; mat 0..2 = branches, 3 = xf
__global__ void adaln_kernel(const float* __restrict__ se,
                             const float* __restrict__ aw, const float* __restrict__ ab,
                             const float* __restrict__ xw, const float* __restrict__ xb,
                             float* __restrict__ scsh) {
    int idx = blockIdx.x * 256 + threadIdx.x;         // 32768 total
    int mat = idx >> 13;
    int rem = idx & 8191;
    int b   = rem >> 11;
    int j   = rem & 2047;
    const float* W;
    float bias;
    if (mat < 3) { W = aw + (size_t)mat * EE * 2048; bias = ab[mat * 2048 + j]; }
    else         { W = xw;                           bias = xb[j]; }
    const float* ser = se + b * EE;
    float acc = bias;
    for (int e = 0; e < EE; ++e) acc += ser[e] * W[(size_t)e * 2048 + j];
    scsh[idx] = acc;
}

// --------------------------- LayerNorm + (1+scale)/shift modulation -> f16
__global__ void ln_mod_kernel(const float* __restrict__ x,
                              const float* __restrict__ scsh,   // already offset by mat
                              _Float16* __restrict__ out) {
    __shared__ float r1[256], r2[256];
    int row = blockIdx.x;                 // 0..2047 (b*512 + t)
    int b   = row >> 9;
    const float* xr = x + (size_t)row * DD;
    float s = 0.f, s2 = 0.f;
    for (int d = threadIdx.x; d < DD; d += 256) { float v = xr[d]; s += v; s2 += v * v; }
    r1[threadIdx.x] = s; r2[threadIdx.x] = s2;
    __syncthreads();
    for (int off = 128; off > 0; off >>= 1) {
        if (threadIdx.x < off) { r1[threadIdx.x] += r1[threadIdx.x + off];
                                 r2[threadIdx.x] += r2[threadIdx.x + off]; }
        __syncthreads();
    }
    float mean = r1[0] * (1.f / DD);
    float var  = r2[0] * (1.f / DD) - mean * mean;
    float rstd = rsqrtf(var + 1e-6f);
    const float* sc = scsh + b * 2048;
    for (int d = threadIdx.x; d < DD; d += 256) {
        float v = (xr[d] - mean) * rstd;
        out[(size_t)row * DD + d] = (_Float16)(v * (1.f + sc[d]) + sc[DD + d]);
    }
}

// ------------- weights: f32 (in,out) -> f16 transposed (out,in) for B-frags
__global__ void wconv_kernel(const float* __restrict__ qw, const float* __restrict__ kw,
                             const float* __restrict__ vw, const float* __restrict__ ow,
                             _Float16* __restrict__ wt) {
    size_t idx = (size_t)blockIdx.x * 256 + threadIdx.x;   // 8M total
    int mat = (int)(idx >> 20);
    size_t rem = idx & ((1u << 20) - 1);
    int k = (int)(rem >> 10), n = (int)(rem & 1023);
    const float* src;
    if (mat < 3)       src = qw + (size_t)mat * 1048576;
    else if (mat == 3) src = kw;
    else if (mat == 4) src = vw;
    else               src = ow + (size_t)(mat - 5) * 1048576;
    wt[(size_t)mat * 1048576 + (size_t)n * 1024 + k] = (_Float16)src[(size_t)k * 1024 + n];
}

// -------------------------------------------------------- WMMA GEMM (K=1024)
// One wave -> 16x64 output tile (4 accumulators reuse A fragment).
// MODE 0: f16 store, batched-row mapping. MODE 1: f16 scatter to Vt(B,H,HD,N).
// MODE 2: f32 store, batched-row mapping.
template <int MODE>
__global__ void __launch_bounds__(128)
gemm16(const _Float16* __restrict__ A, const _Float16* __restrict__ Wt,
       const float* __restrict__ bias, void* __restrict__ Cout,
       int a_base, int a_br, int a_bs,
       int c_base, int c_br, int c_bs) {
    int w  = threadIdx.x >> 5, l = threadIdx.x & 31;
    int lr = l & 15, lh = l >> 4;
    int m0 = blockIdx.x * 16;
    int n0 = (blockIdx.y * 4 + w) * 64;
    int ma = m0 + lr;
    const _Float16* arow = A + (size_t)a_base + (size_t)(ma / a_br) * a_bs
                             + (size_t)(ma % a_br) * DD;
    const _Float16* w0 = Wt + (size_t)(n0 + lr) * DD;
    v8f acc0 = {}, acc1 = {}, acc2 = {}, acc3 = {};
    #pragma unroll 2
    for (int k0 = 0; k0 < DD; k0 += 32) {
        v16h a  = *(const v16h*)(arow + k0 + lh * 16);
        v16h b0 = *(const v16h*)(w0 + k0 + lh * 16);
        v16h b1 = *(const v16h*)(w0 + 16 * DD + k0 + lh * 16);
        v16h b2 = *(const v16h*)(w0 + 32 * DD + k0 + lh * 16);
        v16h b3 = *(const v16h*)(w0 + 48 * DD + k0 + lh * 16);
        acc0 = wmma16(a, b0, acc0);
        acc1 = wmma16(a, b1, acc1);
        acc2 = wmma16(a, b2, acc2);
        acc3 = wmma16(a, b3, acc3);
    }
    v8f accs[4] = {acc0, acc1, acc2, acc3};
    #pragma unroll
    for (int t = 0; t < 4; ++t) {
        #pragma unroll
        for (int r = 0; r < 8; ++r) {
            int m   = m0 + lh * 8 + r;              // C layout: row = r + 8*(lane>>4)
            int col = n0 + t * 16 + lr;
            float val = accs[t][r] + bias[col];
            if (MODE == 2) {
                ((float*)Cout)[(size_t)c_base + (size_t)(m / c_br) * c_bs
                               + (size_t)(m % c_br) * DD + col] = val;
            } else if (MODE == 1) {
                int b = m >> 9, key = m & 511, h = col >> 6, hd = col & 63;
                ((_Float16*)Cout)[(((size_t)(b * HH + h) * HD) + hd) * NN + key] = (_Float16)val;
            } else {
                ((_Float16*)Cout)[(size_t)c_base + (size_t)(m / c_br) * c_bs
                                  + (size_t)(m % c_br) * DD + col] = (_Float16)val;
            }
        }
    }
}

// ----------------------------------------------- Flash attention (per head)
// grid (24, H, B); block 128 = 4 waves; each wave: 16 Q rows x HD=64.
// Transposed formulation: S^T = K*Q^T (rows=keys, cols=queries),
// O^T = V^T*P^T (rows=hd, cols=queries). Per-query softmax stats reduce over
// a lane's own accumulator registers + one shfl_xor(16). exp2-domain softmax.
__global__ void __launch_bounds__(128)
attn_kernel(const _Float16* __restrict__ Q16, const _Float16* __restrict__ K16,
            const _Float16* __restrict__ Vt16, const float* __restrict__ maskf,
            _Float16* __restrict__ AO) {
    __shared__ __align__(16) _Float16 psh[4][16][32];   // [wave][query][key]
    int w  = threadIdx.x >> 5, l = threadIdx.x & 31;
    int lr = l & 15, lh = l >> 4;
    int b = blockIdx.z, h = blockIdx.y;
    int q0 = (blockIdx.x * 4 + w) * 16;

    // Q^T B-fragments: lane = query column (q0+lr), halves = contiguous hd
    const _Float16* qrow = Q16 + ((size_t)(b * T3 + q0 + lr)) * DD + h * HD + lh * 16;
    v16h qb0 = *(const v16h*)(qrow);          // hd 0..31
    v16h qb1 = *(const v16h*)(qrow + 32);     // hd 32..63

    // V^T base: row hd = u*16 + lr, contiguous keys
    const _Float16* vrow = Vt16 + ((size_t)((b * HH + h) * HD + lr)) * NN;
    const float* mrowp = maskf + b * NN;

    float mst = -1e30f, lst = 0.f;            // per-query (lane) running stats
    v8f o0 = {}, o1 = {}, o2 = {}, o3 = {};   // O^T tiles, hd 0..63

    for (int kc = 0; kc < 16; ++kc) {
        int key0 = kc * 32;
        // K A-fragments for the two 16-key tiles (rows = keys)
        const _Float16* kr0 = K16 + ((size_t)(b * NN + key0 + lr)) * DD + h * HD;
        const _Float16* kr1 = kr0 + (size_t)16 * DD;
        v16h ka00 = *(const v16h*)(kr0 + lh * 16);
        v16h ka01 = *(const v16h*)(kr0 + 32 + lh * 16);
        v16h ka10 = *(const v16h*)(kr1 + lh * 16);
        v16h ka11 = *(const v16h*)(kr1 + 32 + lh * 16);
        v8f s0 = {}, s1 = {};
        s0 = wmma16(ka00, qb0, s0); s0 = wmma16(ka01, qb1, s0);
        s1 = wmma16(ka10, qb0, s1); s1 = wmma16(ka11, qb1, s1);

        // additive mask for this lane's 8 keys of each tile (aligned v8f loads)
        v8f madd0 = *(const v8f*)(mrowp + key0 + lh * 8);
        v8f madd1 = *(const v8f*)(mrowp + key0 + 16 + lh * 8);

        float v0[8], v1[8];
        float cm = -1e30f;
        #pragma unroll
        for (int r = 0; r < 8; ++r) {
            float a = fmaf(s0[r], SCL2, madd0[r]);   // logit in exp2 domain + mask
            float c = fmaf(s1[r], SCL2, madd1[r]);
            v0[r] = a; v1[r] = c;
            cm = fmaxf(cm, fmaxf(a, c));
        }
        cm = fmaxf(cm, __shfl_xor(cm, 16, 32));      // combine key halves
        float mnew  = fmaxf(mst, cm);
        float alpha = exp2f(mst - mnew);
        float rs = 0.f;
        v8h p0, p1;
        #pragma unroll
        for (int r = 0; r < 8; ++r) {
            float e0 = exp2f(v0[r] - mnew);
            float e1 = exp2f(v1[r] - mnew);
            rs += e0 + e1;
            p0[r] = (_Float16)e0;
            p1[r] = (_Float16)e1;
        }
        rs += __shfl_xor(rs, 16, 32);
        lst = lst * alpha + rs;
        mst = mnew;
        #pragma unroll
        for (int r = 0; r < 8; ++r) { o0[r] *= alpha; o1[r] *= alpha; o2[r] *= alpha; o3[r] *= alpha; }

        // stage P^T as [query][key]; the B-fragment read is two 16B loads
        *(v8h*)&psh[w][lr][lh * 8]      = p0;   // keys lh*8..+7
        *(v8h*)&psh[w][lr][16 + lh * 8] = p1;   // keys 16+lh*8..+7
        __syncthreads();
        v16h pb = cat8(*(const v8h*)&psh[w][lr][lh * 8], *(const v8h*)&psh[w][lr][16 + lh * 8]);

        const _Float16* va = vrow + key0;
        v16h va0 = cat8(*(const v8h*)(va + (size_t)0 * 16 * NN + lh * 8),
                        *(const v8h*)(va + (size_t)0 * 16 * NN + 16 + lh * 8));
        v16h va1 = cat8(*(const v8h*)(va + (size_t)1 * 16 * NN + lh * 8),
                        *(const v8h*)(va + (size_t)1 * 16 * NN + 16 + lh * 8));
        v16h va2 = cat8(*(const v8h*)(va + (size_t)2 * 16 * NN + lh * 8),
                        *(const v8h*)(va + (size_t)2 * 16 * NN + 16 + lh * 8));
        v16h va3 = cat8(*(const v8h*)(va + (size_t)3 * 16 * NN + lh * 8),
                        *(const v8h*)(va + (size_t)3 * 16 * NN + 16 + lh * 8));
        o0 = wmma16(va0, pb, o0);
        o1 = wmma16(va1, pb, o1);
        o2 = wmma16(va2, pb, o2);
        o3 = wmma16(va3, pb, o3);
        __syncthreads();
    }

    // O^T store: lane = query row (q0+lr); cols hd = u*16 + lh*8 + r (contig 8)
    float inv = 1.f / lst;
    size_t base = ((size_t)(b * T3 + q0 + lr)) * DD + h * HD + lh * 8;
    v8f oo[4] = {o0, o1, o2, o3};
    #pragma unroll
    for (int u = 0; u < 4; ++u) {
        v8h ov;
        #pragma unroll
        for (int r = 0; r < 8; ++r) ov[r] = (_Float16)(oo[u][r] * inv);
        *(v8h*)(AO + base + u * 16) = ov;
    }
}

// ---------------------------------------------------------------------------
extern "C" void kernel_launch(void* const* d_in, const int* in_sizes, int n_in,
                              void* d_out, int out_size, void* d_ws, size_t ws_size,
                              hipStream_t stream) {
    const float* x1   = (const float*)d_in[0];
    const float* x2   = (const float*)d_in[1];
    const float* x3   = (const float*)d_in[2];
    const float* xf   = (const float*)d_in[3];
    const float* emb  = (const float*)d_in[4];
    const unsigned char* mask = (const unsigned char*)d_in[5];  // jnp bool -> 1 byte
    const float* adaln_w    = (const float*)d_in[6];
    const float* adaln_b    = (const float*)d_in[7];
    const float* xf_adaln_w = (const float*)d_in[8];
    const float* xf_adaln_b = (const float*)d_in[9];
    const float* q_w   = (const float*)d_in[10];
    const float* q_b   = (const float*)d_in[11];
    const float* k_w   = (const float*)d_in[12];
    const float* k_b   = (const float*)d_in[13];
    const float* v_w   = (const float*)d_in[14];
    const float* v_b   = (const float*)d_in[15];
    const float* out_w = (const float*)d_in[16];
    const float* out_b = (const float*)d_in[17];

    // Workspace layout (~64.2 MB)
    char* ws = (char*)d_ws;
    size_t off = 0;
    float* se    = (float*)(ws + off); off += 4096 * 4;
    float* maskf = (float*)(ws + off); off += 2048 * 4;
    float* scsh  = (float*)(ws + off); off += 32768 * 4;
    _Float16* wt16 = (_Float16*)(ws + off); off += 8ull * 1048576 * 2;   // 8 transposed weights
    _Float16* h16  = (_Float16*)(ws + off); off += 3ull * 2048 * DD * 2; // modulated branches
    _Float16* xf16 = (_Float16*)(ws + off); off += 2048ull * DD * 2;
    _Float16* q16  = (_Float16*)(ws + off); off += (size_t)BB * T3 * DD * 2; // (B,3T,D)
    _Float16* k16  = (_Float16*)(ws + off); off += 2048ull * DD * 2;         // (B*N,D)
    _Float16* vt16 = (_Float16*)(ws + off); off += (size_t)BB * HH * HD * NN * 2; // (B,H,HD,N)
    _Float16* ao16 = (_Float16*)(ws + off); off += (size_t)BB * T3 * DD * 2; // (B,3T,D)

    // 1) silu(emb) + additive mask
    silu_kernel<<<16, 256, 0, stream>>>(emb, se);
    maskf_kernel<<<8, 256, 0, stream>>>(mask, maskf);
    // 2) AdaLN scale/shift (4 mats x 4 batch x 2048)
    adaln_kernel<<<128, 256, 0, stream>>>(se, adaln_w, adaln_b, xf_adaln_w, xf_adaln_b, scsh);
    // 3) weights -> f16 transposed
    wconv_kernel<<<32768, 256, 0, stream>>>(q_w, k_w, v_w, out_w, wt16);
    // 4) LN + modulation -> f16
    const float* xs[3] = {x1, x2, x3};
    for (int i = 0; i < 3; ++i)
        ln_mod_kernel<<<2048, 256, 0, stream>>>(xs[i], scsh + i * 8192,
                                                h16 + (size_t)i * 2048 * DD);
    ln_mod_kernel<<<2048, 256, 0, stream>>>(xf, scsh + 3 * 8192, xf16);

    dim3 ggrid(128, 4), gblk(128);
    // 5) Q projections (3 branches) -> q16 (B,3T,D)
    for (int i = 0; i < 3; ++i)
        gemm16<0><<<ggrid, gblk, 0, stream>>>(h16 + (size_t)i * 2048 * DD,
                                              wt16 + (size_t)i * 1048576,
                                              q_b + i * DD, q16,
                                              0, 2048, 0,
                                              i * TT * DD, TT, T3 * DD);
    // 6) K projection -> k16 (B*N,D)
    gemm16<0><<<ggrid, gblk, 0, stream>>>(xf16, wt16 + 3ull * 1048576, k_b, k16,
                                          0, 2048, 0, 0, 2048, 0);
    // 7) V projection -> vt16 (B,H,HD,N) scatter
    gemm16<1><<<ggrid, gblk, 0, stream>>>(xf16, wt16 + 4ull * 1048576, v_b, vt16,
                                          0, 2048, 0, 0, 2048, 0);
    // 8) attention
    attn_kernel<<<dim3(24, HH, BB), 128, 0, stream>>>(q16, k16, vt16, maskf, ao16);
    // 9) out projections -> d_out (f32, concat of 3 branches)
    for (int i = 0; i < 3; ++i)
        gemm16<2><<<ggrid, gblk, 0, stream>>>(ao16, wt16 + (size_t)(5 + i) * 1048576,
                                              out_b + i * DD, d_out,
                                              i * TT * DD, TT, T3 * DD,
                                              i * 2048 * DD, 2048, 0);
    (void)in_sizes; (void)n_in; (void)out_size; (void)ws_size;
}
